// KVAttention_41455024341295
// MI455X (gfx1250) — matrix operand order repaired and checked by reading, not compiled, not run on hardware
//
#include <hip/hip_runtime.h>

// ---------------------------------------------------------------------------
// Fused causal attention for MI455X (gfx1250, wave32, WMMA).
//   ref: q=xWq^T+bq, k=xWk^T+bk, v=xWv^T+bv; softmax(mask(qk)/sqrt(E)) v
// f16 WMMA (16x16x32, f32 accum); one-time fp32->f16 staging; flash online
// softmax; causal tile skipping; V transposed; double-buffered fragment
// pipeline; M=32 register blocking so every B fragment feeds 2 WMMAs
// (halves L2 traffic per FLOP vs M=16 blocking).
// ---------------------------------------------------------------------------

typedef _Float16 half_t;
typedef __attribute__((ext_vector_type(16))) _Float16 v16h;
typedef __attribute__((ext_vector_type(8)))  _Float16 v8h;
typedef __attribute__((ext_vector_type(8)))  float    v8f;
typedef __attribute__((ext_vector_type(4)))  float    v4f;

#define EMB   512
#define LSEQ  4096
#define NBAT  4
#define BC    128
#define BR    32
#define QKSCALE 0.044194173824159216f   // 1/sqrt(512)

static __device__ inline v16h combine16(v8h lo, v8h hi) {
  v16h r;
#pragma unroll
  for (int i = 0; i < 8; ++i) { r[i] = lo[i]; r[8 + i] = hi[i]; }
  return r;
}

// ---------------------------------------------------------------------------
// Kernel 0: fp32 -> f16 bulk conversion (x and the three W matrices).
// ---------------------------------------------------------------------------
__global__ __launch_bounds__(256)
void cvt_f32_to_f16_kernel(const float* __restrict__ src,
                           half_t* __restrict__ dst, int n)
{
  const int i = (blockIdx.x * 256 + threadIdx.x) * 8;
  if (i >= n) return;
  v4f f0 = *(const v4f*)(src + i);
  v4f f1 = *(const v4f*)(src + i + 4);
  v8h h;
#pragma unroll
  for (int j = 0; j < 4; ++j) { h[j] = (half_t)f0[j]; h[4 + j] = (half_t)f1[j]; }
  *(v8h*)(dst + i) = h;
}

// ---------------------------------------------------------------------------
// Kernel 1: QKV projection, M=32 register blocking, double-buffered.
// grid = (BL/32, 3), block = 256 (8 waves); wave owns 64 output columns
// for 32 rows.  Each B (W) fragment feeds 2 WMMAs.
// ---------------------------------------------------------------------------
__global__ __launch_bounds__(256)
void qkv_proj_kernel(const half_t* __restrict__ xh,
                     const half_t* __restrict__ Wqh, const float* __restrict__ bq,
                     const half_t* __restrict__ Wkh, const float* __restrict__ bk,
                     const half_t* __restrict__ Wvh, const float* __restrict__ bv,
                     half_t* __restrict__ Qh, half_t* __restrict__ Kh,
                     half_t* __restrict__ Vt)
{
  const int which = blockIdx.y;
  const half_t* W   = (which == 0) ? Wqh : (which == 1) ? Wkh : Wvh;
  const float* bias = (which == 0) ? bq  : (which == 1) ? bk  : bv;

  const int m0    = blockIdx.x * 32;
  const int wave  = threadIdx.x >> 5;
  const int lane  = threadIdx.x & 31;
  const int lh    = lane >> 4;
  const int ll    = lane & 15;
  const int nbase = wave * 64;

  v8f acc[2][4] = {};

  const half_t* pa[2];
#pragma unroll
  for (int t = 0; t < 2; ++t)
    pa[t] = xh + (size_t)(m0 + t * 16 + ll) * EMB + lh * 8;
  const half_t* pb[4];
#pragma unroll
  for (int nt = 0; nt < 4; ++nt)
    pb[nt] = W + (size_t)(nbase + nt * 16 + ll) * EMB + lh * 8;

  // double-buffered fragments (constant indices after full unroll)
  v8h A_lo[2][2], A_hi[2][2], B_lo[2][4], B_hi[2][4];

#define PROJ_LOAD(st, k0)                                                  \
  do {                                                                     \
    _Pragma("unroll")                                                      \
    for (int t = 0; t < 2; ++t) {                                          \
      A_lo[st][t] = *(const v8h*)(pa[t] + (k0));                           \
      A_hi[st][t] = *(const v8h*)(pa[t] + (k0) + 16);                      \
    }                                                                      \
    _Pragma("unroll")                                                      \
    for (int nt = 0; nt < 4; ++nt) {                                       \
      B_lo[st][nt] = *(const v8h*)(pb[nt] + (k0));                         \
      B_hi[st][nt] = *(const v8h*)(pb[nt] + (k0) + 16);                    \
    }                                                                      \
  } while (0)

  PROJ_LOAD(0, 0);
#pragma unroll
  for (int it = 0; it < EMB / 32; ++it) {
    const int cur = it & 1;
    if (it + 1 < EMB / 32) PROJ_LOAD(cur ^ 1, (it + 1) * 32);  // prefetch
    v16h a0 = combine16(A_lo[cur][0], A_hi[cur][0]);
    v16h a1 = combine16(A_lo[cur][1], A_hi[cur][1]);
#pragma unroll
    for (int nt = 0; nt < 4; ++nt) {
      v16h bf = combine16(B_lo[cur][nt], B_hi[cur][nt]);
      acc[0][nt] = __builtin_amdgcn_wmma_f32_16x16x32_f16(
          false, a0, false, bf, (short)0, acc[0][nt], false, false);
      acc[1][nt] = __builtin_amdgcn_wmma_f32_16x16x32_f16(
          false, a1, false, bf, (short)0, acc[1][nt], false, false);
    }
  }
#undef PROJ_LOAD

  // Epilogue: bias add, f16 convert, store.
#pragma unroll
  for (int nt = 0; nt < 4; ++nt) {
    const int n = nbase + nt * 16 + ll;
    const float bn = bias[n];
#pragma unroll
    for (int t = 0; t < 2; ++t) {
#pragma unroll
      for (int r = 0; r < 8; ++r) {
        const int mm = m0 + t * 16 + r + lh * 8;   // C/D lane layout
        half_t hv = (half_t)(acc[t][nt][r] + bn);
        if (which == 0) {
          Qh[(size_t)mm * EMB + n] = hv;
        } else if (which == 1) {
          Kh[(size_t)mm * EMB + n] = hv;
        } else {
          const int b = mm >> 12;                  // / LSEQ
          const int l = mm & (LSEQ - 1);
          Vt[((size_t)b * EMB + n) * LSEQ + l] = hv;
        }
      }
    }
  }
}

// ---------------------------------------------------------------------------
// Kernel 2: causal flash attention, Br=32.  grid = (L/32, B), block = 256.
// Wave w computes S columns [w*16, w*16+16) for both q-tiles (K frag reused),
// then accumulates O columns [w*64, w*64+64) for both q-tiles (V frag reused).
// ---------------------------------------------------------------------------
__global__ __launch_bounds__(256)
void flash_attn_kernel(const half_t* __restrict__ Qh,
                       const half_t* __restrict__ Kh,
                       const half_t* __restrict__ Vt,
                       float* __restrict__ out)
{
  __shared__ half_t sQ[BR * EMB];            // 32 KB
  __shared__ float  sS[BR * BC];             // 16 KB
  __shared__ half_t sP[BR * BC];             // 8 KB
  __shared__ float  sRed[BR * 8];            // 1 KB
  __shared__ float  row_m[BR], row_l[BR], row_scale[BR], tile_max[BR];

  const int b    = blockIdx.y;
  const int q0   = blockIdx.x * BR;
  const int tid  = threadIdx.x;
  const int wave = tid >> 5;
  const int lane = tid & 31;
  const int lh   = lane >> 4;
  const int ll   = lane & 15;

  // Stage Q tile (32 x 512 f16) into LDS: 64 halves per thread.
  {
    const half_t* src = Qh + (size_t)(b * LSEQ + q0) * EMB;
    const int base = tid * 64;
#pragma unroll
    for (int e = 0; e < 64; e += 8)
      *(v8h*)(sQ + base + e) = *(const v8h*)(src + base + e);
  }
  if (tid < BR) { row_m[tid] = -3.0e38f; row_l[tid] = 0.0f; }
  __syncthreads();

  v8f oacc[2][4] = {};
  const int ebase = wave * 64;

  for (int j0 = 0; j0 < q0 + BR; j0 += BC) { // causal: skip tiles past q-range
    // ---- Phase 1: S = Q K^T for both q-tiles, K frag reused ---------------
    v8f s[2] = {};
    const int key  = j0 + wave * 16 + ll;
    const int krow = (key < LSEQ) ? key : (LSEQ - 1);
    const half_t* kbase = Kh + (size_t)(b * LSEQ + krow) * EMB + lh * 8;
    const half_t* qbase0 = sQ + (0 * 16 + ll) * EMB + lh * 8;
    const half_t* qbase1 = sQ + (1 * 16 + ll) * EMB + lh * 8;
    {
      v8h A_lo[2][2], A_hi[2][2], Bk_lo[2], Bk_hi[2];
#define S_LOAD(st, k0)                                                     \
      do {                                                                 \
        A_lo[st][0] = *(const v8h*)(qbase0 + (k0));                        \
        A_hi[st][0] = *(const v8h*)(qbase0 + (k0) + 16);                   \
        A_lo[st][1] = *(const v8h*)(qbase1 + (k0));                        \
        A_hi[st][1] = *(const v8h*)(qbase1 + (k0) + 16);                   \
        Bk_lo[st]   = *(const v8h*)(kbase + (k0));                         \
        Bk_hi[st]   = *(const v8h*)(kbase + (k0) + 16);                    \
      } while (0)
      S_LOAD(0, 0);
#pragma unroll
      for (int it = 0; it < EMB / 32; ++it) {
        const int cur = it & 1;
        if (it + 1 < EMB / 32) S_LOAD(cur ^ 1, (it + 1) * 32);  // prefetch
        v16h bf = combine16(Bk_lo[cur], Bk_hi[cur]);
        s[0] = __builtin_amdgcn_wmma_f32_16x16x32_f16(
            false, combine16(A_lo[cur][0], A_hi[cur][0]), false, bf,
            (short)0, s[0], false, false);
        s[1] = __builtin_amdgcn_wmma_f32_16x16x32_f16(
            false, combine16(A_lo[cur][1], A_hi[cur][1]), false, bf,
            (short)0, s[1], false, false);
      }
#undef S_LOAD
    }
    // mask (before scaling, as in reference) and dump to LDS
#pragma unroll
    for (int t = 0; t < 2; ++t) {
#pragma unroll
      for (int r = 0; r < 8; ++r) {
        const int row  = t * 16 + r + lh * 8;
        const int qrow = q0 + row;
        const float raw = (key > qrow) ? -1e9f : s[t][r];
        sS[row * BC + wave * 16 + ll] = raw * QKSCALE;
      }
    }
    __syncthreads();

    // ---- Phase 2: block-cooperative online softmax (32 rows) --------------
    const int srow = tid >> 3;               // 32 rows
    const int schk = tid & 7;                // 8 chunks of 16 cols
    {
      const float* sr = sS + srow * BC + schk * 16;
      float mx = sr[0];
#pragma unroll
      for (int i = 1; i < 16; ++i) mx = fmaxf(mx, sr[i]);
      sRed[srow * 8 + schk] = mx;
    }
    __syncthreads();
    if (tid < BR) {
      float mx = row_m[tid];
#pragma unroll
      for (int i = 0; i < 8; ++i) mx = fmaxf(mx, sRed[tid * 8 + i]);
      tile_max[tid] = mx;
    }
    __syncthreads();
    {
      const float mx = tile_max[srow];
      const float* sr = sS + srow * BC + schk * 16;
      half_t* pp = sP + srow * BC + schk * 16;
      float sum = 0.0f;
#pragma unroll
      for (int i = 0; i < 16; ++i) {
        const float p = __expf(sr[i] - mx);
        sum += p;
        pp[i] = (half_t)p;
      }
      sRed[srow * 8 + schk] = sum;
    }
    __syncthreads();
    if (tid < BR) {
      float sum = 0.0f;
#pragma unroll
      for (int i = 0; i < 8; ++i) sum += sRed[tid * 8 + i];
      const float mnew = tile_max[tid];
      const float fac  = __expf(row_m[tid] - mnew);
      row_l[tid]     = row_l[tid] * fac + sum;
      row_m[tid]     = mnew;
      row_scale[tid] = fac;
    }
    __syncthreads();

    // ---- Phase 3: rescale O, accumulate O += P V (V frag reused) ----------
#pragma unroll
    for (int t = 0; t < 2; ++t) {
      v4f f0 = *(const v4f*)(row_scale + t * 16 + lh * 8);
      v4f f1 = *(const v4f*)(row_scale + t * 16 + lh * 8 + 4);
#pragma unroll
      for (int et = 0; et < 4; ++et)
#pragma unroll
        for (int r = 0; r < 4; ++r) {
          oacc[t][et][r]     *= f0[r];
          oacc[t][et][4 + r] *= f1[r];
        }
    }
    {
      const half_t* vtbase = Vt + (size_t)b * EMB * LSEQ;
      v8h P_lo[2][2], P_hi[2][2], Vb_lo[2][4], Vb_hi[2][4];
#define PV_LOAD(st, kk)                                                    \
      do {                                                                 \
        _Pragma("unroll")                                                  \
        for (int t = 0; t < 2; ++t) {                                      \
          const half_t* pa2 = sP + (t * 16 + ll) * BC + (kk) + lh * 16;    \
          P_lo[st][t] = *(const v8h*)(pa2);                                \
          P_hi[st][t] = *(const v8h*)(pa2 + 8);                            \
        }                                                                  \
        int kst = j0 + (kk) + lh * 16;                                     \
        if (kst > LSEQ - 16) kst = LSEQ - 16;  /* clamp; P==0 there */     \
        _Pragma("unroll")                                                  \
        for (int et = 0; et < 4; ++et) {                                   \
          const half_t* pv =                                               \
              vtbase + (size_t)(ebase + et * 16 + ll) * LSEQ + kst;        \
          Vb_lo[st][et] = *(const v8h*)(pv);                               \
          Vb_hi[st][et] = *(const v8h*)(pv + 8);                           \
        }                                                                  \
      } while (0)
      PV_LOAD(0, 0);
#pragma unroll
      for (int it = 0; it < BC / 32; ++it) {
        const int cur = it & 1;
        if (it + 1 < BC / 32) PV_LOAD(cur ^ 1, (it + 1) * 32);  // prefetch
        v16h a0 = combine16(P_lo[cur][0], P_hi[cur][0]);
        v16h a1 = combine16(P_lo[cur][1], P_hi[cur][1]);
#pragma unroll
        for (int et = 0; et < 4; ++et) {
          v16h bf = combine16(Vb_lo[cur][et], Vb_hi[cur][et]);
          oacc[0][et] = __builtin_amdgcn_wmma_f32_16x16x32_f16(
              false, a0, false, bf, (short)0, oacc[0][et], false, false);
          oacc[1][et] = __builtin_amdgcn_wmma_f32_16x16x32_f16(
              false, a1, false, bf, (short)0, oacc[1][et], false, false);
        }
      }
#undef PV_LOAD
    }
    __syncthreads();                         // protect sS/sP reuse
  }

  // ---- Finalize: O /= l, store fp32 ---------------------------------------
#pragma unroll
  for (int t = 0; t < 2; ++t) {
    v4f l0 = *(const v4f*)(row_l + t * 16 + lh * 8);
    v4f l1 = *(const v4f*)(row_l + t * 16 + lh * 8 + 4);
#pragma unroll
    for (int et = 0; et < 4; ++et) {
      const int e = ebase + et * 16 + ll;
#pragma unroll
      for (int r = 0; r < 8; ++r) {
        const float denom = (r < 4) ? l0[r] : l1[r - 4];
        const int qrow = q0 + t * 16 + r + lh * 8;
        out[(size_t)(b * LSEQ + qrow) * EMB + e] = oacc[t][et][r] / denom;
      }
    }
  }
}

// ---------------------------------------------------------------------------
extern "C" void kernel_launch(void* const* d_in, const int* in_sizes, int n_in,
                              void* d_out, int out_size, void* d_ws, size_t ws_size,
                              hipStream_t stream) {
  const float* x  = (const float*)d_in[0];
  const float* Wq = (const float*)d_in[1];
  const float* bq = (const float*)d_in[2];
  const float* Wk = (const float*)d_in[3];
  const float* bk = (const float*)d_in[4];
  const float* Wv = (const float*)d_in[5];
  const float* bv = (const float*)d_in[6];
  float* out = (float*)d_out;

  const size_t nElems = (size_t)NBAT * LSEQ * EMB;     // 8M elements
  const size_t nW     = (size_t)EMB * EMB;             // 256K elements
  half_t* Qh  = (half_t*)d_ws;                         // 16 MB
  half_t* Kh  = Qh + nElems;                           // 16 MB
  half_t* Vt  = Kh + nElems;                           // 16 MB (transposed)
  half_t* xh  = Vt + nElems;                           // 16 MB
  half_t* Wqh = xh + nElems;                           // 0.5 MB
  half_t* Wkh = Wqh + nW;                              // 0.5 MB
  half_t* Wvh = Wkh + nW;                              // 0.5 MB

  // 0) one-time fp32 -> f16 staging
  cvt_f32_to_f16_kernel<<<(int)(nElems / 8 / 256), 256, 0, stream>>>(x, xh, (int)nElems);
  cvt_f32_to_f16_kernel<<<(int)(nW / 8 / 256), 256, 0, stream>>>(Wq, Wqh, (int)nW);
  cvt_f32_to_f16_kernel<<<(int)(nW / 8 / 256), 256, 0, stream>>>(Wk, Wkh, (int)nW);
  cvt_f32_to_f16_kernel<<<(int)(nW / 8 / 256), 256, 0, stream>>>(Wv, Wvh, (int)nW);

  // 1) QKV projection (M=32 blocking)
  dim3 g1((NBAT * LSEQ) / 32, 3);
  qkv_proj_kernel<<<g1, 256, 0, stream>>>(xh, Wqh, bq, Wkh, bk, Wvh, bv,
                                          Qh, Kh, Vt);
  // 2) causal flash attention (Br=32)
  dim3 g2(LSEQ / 32, NBAT);
  flash_attn_kernel<<<g2, 256, 0, stream>>>(Qh, Kh, Vt, out);
}
